// Attention_49606872268904
// MI455X (gfx1250) — matrix operand order repaired and checked by reading, not compiled, not run on hardware
//
#include <hip/hip_runtime.h>
#include <hip/hip_bf16.h>

// ---------------- problem constants ----------------
#define S_LEN 2048
#define DIM   4096
#define NH    32
#define NKV   8
#define HD    128
#define QDIM  (NH * HD)    // 4096
#define KVDIM (NKV * HD)   // 1024

typedef __bf16 bf16_t;
typedef __attribute__((ext_vector_type(16))) __bf16 v16bf;
typedef __attribute__((ext_vector_type(8)))  __bf16 v8bf;
typedef __attribute__((ext_vector_type(4)))  __bf16 v4bf;
typedef __attribute__((ext_vector_type(8)))  float  v8f;
typedef __attribute__((ext_vector_type(4)))  float  v4f;

// =====================================================================
// fp32 -> bf16 bulk conversion (one-time cost; halves streamed bytes and
// removes all cvt work from the GEMM hot loops)
// =====================================================================
__global__ void conv_f32_bf16(const float* __restrict__ in,
                              bf16_t* __restrict__ out, int n4)
{
  int i = blockIdx.x * blockDim.x + threadIdx.x;
  if (i >= n4) return;
  v4f v = *(const v4f*)(in + 4 * (size_t)i);
  v4bf o;
  o[0] = (__bf16)v[0]; o[1] = (__bf16)v[1];
  o[2] = (__bf16)v[2]; o[3] = (__bf16)v[3];
  *(v4bf*)(out + 4 * (size_t)i) = o;
}

// =====================================================================
// GEMM:  C[M,N] = A[M,K] * W[N,K]^T   (bf16 operands, fp32 out)
// LDS-tiled: block = 256 threads (8 waves), block tile 128x128, K-step 32.
// Wave (mw,nw) -> 32x64 sub-tile = 2x4 WMMA accumulators.
// grid = (N/128, M/128).
// =====================================================================
#define LDSTRIDE 40   // bf16 elems per row: 80B rows keep 16B alignment
__global__ __launch_bounds__(256) void gemm_bf16_wmma(
    const bf16_t* __restrict__ A, const bf16_t* __restrict__ W,
    float* __restrict__ C, int M, int N, int K)
{
  __shared__ bf16_t As[128][LDSTRIDE];
  __shared__ bf16_t Bs[128][LDSTRIDE];

  const int tid  = threadIdx.x;
  const int lane = tid & 31;
  const int wave = tid >> 5;
  const int mw   = wave & 3;      // 0..3 : 32-row group
  const int nw   = wave >> 2;     // 0..1 : 64-col group
  const int m    = lane & 15;
  const int hi   = lane >> 4;
  const int row0 = blockIdx.y * 128;
  const int col0 = blockIdx.x * 128;

  const int ldrow   = tid >> 1;          // 0..127
  const int ldchunk = (tid & 1) * 16;    // 0 or 16

  v8f acc[2][4];
#pragma unroll
  for (int mt = 0; mt < 2; ++mt)
#pragma unroll
    for (int nt = 0; nt < 4; ++nt) acc[mt][nt] = (v8f){};

  for (int k0 = 0; k0 < K; k0 += 32) {
    // ---- stage 128x32 A-slab and 128x32 B-slab into LDS ----
    *(v16bf*)&As[ldrow][ldchunk] =
        *(const v16bf*)(A + (size_t)(row0 + ldrow) * K + k0 + ldchunk);
    *(v16bf*)&Bs[ldrow][ldchunk] =
        *(const v16bf*)(W + (size_t)(col0 + ldrow) * K + k0 + ldchunk);
    __builtin_prefetch(A + (size_t)(row0 + ldrow) * K + k0 + 32, 0, 0);
    __builtin_prefetch(W + (size_t)(col0 + ldrow) * K + k0 + 32, 0, 0);
    __syncthreads();

    // ---- A fragments (16-bit A 16x32 layout) ----
    v16bf afrag[2];
#pragma unroll
    for (int mt = 0; mt < 2; ++mt) {
      const int arow = mw * 32 + mt * 16 + m;
      v8bf lo = *(const v8bf*)&As[arow][16 * hi];       // K = 16*hi + e
      v8bf up = *(const v8bf*)&As[arow][16 * hi + 8];   // K = 16*hi + 8 + e
      afrag[mt] = __builtin_shufflevector(lo, up,
          0, 1, 2, 3, 4, 5, 6, 7, 8, 9, 10, 11, 12, 13, 14, 15);
    }
    // ---- B fragments + 8 WMMAs ----
#pragma unroll
    for (int nt = 0; nt < 4; ++nt) {
      const int bn = nw * 64 + nt * 16 + m;
      v16bf b = *(const v16bf*)&Bs[bn][16 * hi];        // K = 16*hi + e
      acc[0][nt] = __builtin_amdgcn_wmma_f32_16x16x32_bf16(
          false, afrag[0], false, b, (short)0, acc[0][nt], false, false);
      acc[1][nt] = __builtin_amdgcn_wmma_f32_16x16x32_bf16(
          false, afrag[1], false, b, (short)0, acc[1][nt], false, false);
    }
    __syncthreads();
  }

  // ---- C/D layout: VGPR r -> row r (+8 for upper lanes), col = lane%16 ----
#pragma unroll
  for (int mt = 0; mt < 2; ++mt)
#pragma unroll
    for (int nt = 0; nt < 4; ++nt)
#pragma unroll
      for (int r = 0; r < 8; ++r) {
        const size_t row = (size_t)(row0 + mw * 32 + mt * 16 + r + 8 * hi);
        C[row * N + col0 + nw * 64 + nt * 16 + m] = acc[mt][nt][r];
      }
}

// =====================================================================
// RoPE + bf16 conversion for Q (scale folded in) and K.
// =====================================================================
__global__ void rope_conv_kernel(const float* __restrict__ tf,
                                 const float* __restrict__ cosb,
                                 const float* __restrict__ sinb,
                                 bf16_t* __restrict__ tbf,
                                 int nheads, float scale)
{
  int idx = blockIdx.x * blockDim.x + threadIdx.x;  // < S * nheads * 64
  int i = idx & 63;
  int t = idx >> 6;
  int h = t % nheads;
  int s = t / nheads;
  if (s >= S_LEN) return;
  const float c  = cosb[s * 64 + i];
  const float sn = sinb[s * 64 + i];
  const size_t base = ((size_t)s * nheads + h) * HD + 2 * i;
  const float tr = tf[base];
  const float ti = tf[base + 1];
  tbf[base]     = (__bf16)((tr * c - ti * sn) * scale);
  tbf[base + 1] = (__bf16)((tr * sn + ti * c) * scale);
}

// V: fp32 [S][NKV*HD] -> bf16 transposed [NKV][HD][S]  (B-operand friendly)
__global__ void conv_v_kernel(const float* __restrict__ vf, bf16_t* __restrict__ vt)
{
  int idx = blockIdx.x * blockDim.x + threadIdx.x;  // < S * NKV * HD
  int d = idx & (HD - 1);
  int t = idx >> 7;
  int h = t & (NKV - 1);
  int s = t >> 3;
  if (s >= S_LEN) return;
  vt[((size_t)h * HD + d) * S_LEN + s] = (__bf16)vf[((size_t)s * NKV + h) * HD + d];
}

// =====================================================================
// Flash attention: one wave per (16-query tile, head). Causal.
// S = Q*K^T via WMMA; online softmax fp32; P staged C->A layout via LDS;
// O += P*V via WMMA with V pre-transposed. Output written as bf16.
// grid = (S/16, NH), block = 32.
// =====================================================================
__global__ __launch_bounds__(32) void flash_kernel(
    const bf16_t* __restrict__ qbf,   // [S][NH*HD], scale pre-applied
    const bf16_t* __restrict__ kbf,   // [S][NKV*HD]
    const bf16_t* __restrict__ vt,    // [NKV][HD][S]
    bf16_t* __restrict__ attn)        // [S][NH*HD] bf16
{
  __shared__ float pbuf[16][34];

  const int lane = threadIdx.x & 31;
  const int m    = lane & 15;
  const int hi   = lane >> 4;
  const int q0   = blockIdx.x * 16;
  const int h    = blockIdx.y;
  const int kvh  = h >> 2;           // REPEATS = 4

  // ---- Q fragments: 4 x (16x32 over head-dim) in A layout ----
  v16bf qa[4];
  const bf16_t* qrow = qbf + (size_t)(q0 + m) * QDIM + h * HD;
#pragma unroll
  for (int c = 0; c < 4; ++c) {
    v8bf lo = *(const v8bf*)(qrow + 32 * c + 16 * hi);
    v8bf up = *(const v8bf*)(qrow + 32 * c + 16 * hi + 8);
    qa[c] = __builtin_shufflevector(lo, up,
        0, 1, 2, 3, 4, 5, 6, 7, 8, 9, 10, 11, 12, 13, 14, 15);
  }

  v8f o[8];
#pragma unroll
  for (int t = 0; t < 8; ++t) o[t] = (v8f){};
  v8f mrow, lrow;
#pragma unroll
  for (int r = 0; r < 8; ++r) { mrow[r] = -3.0e38f; lrow[r] = 0.0f; }

  const int npair = (q0 + 47) >> 5;   // 32-key pairs covering keys 0..q0+15
  for (int p = 0; p < npair; ++p) {
    const int kb = p * 32;

    // ---- S tiles: keys kb..kb+15 and kb+16..kb+31 ----
    v8f s0 = {}, s1 = {};
#pragma unroll
    for (int c = 0; c < 4; ++c) {
      const bf16_t* kr0 = kbf + (size_t)(kb + m) * KVDIM      + kvh * HD + 32 * c + 16 * hi;
      const bf16_t* kr1 = kbf + (size_t)(kb + 16 + m) * KVDIM + kvh * HD + 32 * c + 16 * hi;
      v16bf b0 = *(const v16bf*)kr0;
      v16bf b1 = *(const v16bf*)kr1;
      s0 = __builtin_amdgcn_wmma_f32_16x16x32_bf16(false, qa[c], false, b0, (short)0, s0, false, false);
      s1 = __builtin_amdgcn_wmma_f32_16x16x32_bf16(false, qa[c], false, b1, (short)0, s1, false, false);
    }

    // ---- causal mask (row = q0 + r + 8*hi, col = kb(+16) + m) ----
#pragma unroll
    for (int r = 0; r < 8; ++r) {
      const int qi = q0 + r + 8 * hi;
      if (kb + m > qi)      s0[r] = -1.0e30f;
      if (kb + 16 + m > qi) s1[r] = -1.0e30f;
    }

    // ---- online softmax stats (reduce across 16-lane column groups) ----
    v8f alpha;
#pragma unroll
    for (int r = 0; r < 8; ++r) {
      float v = fmaxf(s0[r], s1[r]);
#pragma unroll
      for (int off = 1; off < 16; off <<= 1)
        v = fmaxf(v, __shfl_xor(v, off, 16));
      const float nm = fmaxf(mrow[r], v);
      alpha[r] = __expf(mrow[r] - nm);
      s0[r] = __expf(s0[r] - nm);
      s1[r] = __expf(s1[r] - nm);
      float rs = s0[r] + s1[r];
#pragma unroll
      for (int off = 1; off < 16; off <<= 1)
        rs += __shfl_xor(rs, off, 16);
      lrow[r] = lrow[r] * alpha[r] + rs;
      mrow[r] = nm;
    }

    // ---- stage P: C layout -> LDS -> A layout (16x32, K = key offset) ----
#pragma unroll
    for (int r = 0; r < 8; ++r) {
      pbuf[r + 8 * hi][m]      = s0[r];
      pbuf[r + 8 * hi][m + 16] = s1[r];
    }
    asm volatile("s_wait_dscnt 0" ::: "memory");
    v16bf pa;
#pragma unroll
    for (int e = 0; e < 8; ++e) {
      pa[e]     = (__bf16)pbuf[m][16 * hi + e];
      pa[8 + e] = (__bf16)pbuf[m][16 * hi + 8 + e];
    }
    asm volatile("s_wait_dscnt 0" ::: "memory");

    // ---- O = O*alpha + P*V ----
#pragma unroll
    for (int t = 0; t < 8; ++t) {
#pragma unroll
      for (int r = 0; r < 8; ++r) o[t][r] *= alpha[r];
      const bf16_t* vp = vt + ((size_t)kvh * HD + 16 * t + m) * S_LEN + kb + 16 * hi;
      v16bf vb = *(const v16bf*)vp;
      o[t] = __builtin_amdgcn_wmma_f32_16x16x32_bf16(false, pa, false, vb, (short)0, o[t], false, false);
    }
  }

  // ---- normalize and store bf16 attention output ----
#pragma unroll
  for (int t = 0; t < 8; ++t) {
#pragma unroll
    for (int r = 0; r < 8; ++r) {
      const float val = o[t][r] / lrow[r];
      attn[(size_t)(q0 + r + 8 * hi) * QDIM + h * HD + 16 * t + m] = (__bf16)val;
    }
  }
}

// =====================================================================
// Host-side launcher
// =====================================================================
extern "C" void kernel_launch(void* const* d_in, const int* in_sizes, int n_in,
                              void* d_out, int out_size, void* d_ws, size_t ws_size,
                              hipStream_t stream)
{
  (void)in_sizes; (void)n_in; (void)out_size; (void)ws_size;

  const float* x    = (const float*)d_in[0];   // [S, DIM]
  const float* wq   = (const float*)d_in[1];   // [QDIM, DIM]
  const float* wk   = (const float*)d_in[2];   // [KVDIM, DIM]
  const float* wv   = (const float*)d_in[3];   // [KVDIM, DIM]
  const float* wo   = (const float*)d_in[4];   // [DIM, QDIM]
  const float* fcos = (const float*)d_in[5];   // [S, HD/2]
  const float* fsin = (const float*)d_in[6];   // [S, HD/2]
  // d_in[7..10]: cache_k, cache_v, positions, mask — not needed here

  // ---- workspace carve-up ----
  char* ws = (char*)d_ws;
  bf16_t* xbf  = (bf16_t*)ws;  ws += (size_t)S_LEN * DIM   * sizeof(bf16_t);  // 16 MB
  bf16_t* wqb  = (bf16_t*)ws;  ws += (size_t)QDIM  * DIM   * sizeof(bf16_t);  // 32 MB
  bf16_t* wkb  = (bf16_t*)ws;  ws += (size_t)KVDIM * DIM   * sizeof(bf16_t);  //  8 MB
  bf16_t* wvb  = (bf16_t*)ws;  ws += (size_t)KVDIM * DIM   * sizeof(bf16_t);  //  8 MB
  bf16_t* wob  = (bf16_t*)ws;  ws += (size_t)DIM   * QDIM  * sizeof(bf16_t);  // 32 MB
  float*  qf   = (float*)ws;   ws += (size_t)S_LEN * QDIM  * sizeof(float);   // 32 MB
  float*  kf   = (float*)ws;   ws += (size_t)S_LEN * KVDIM * sizeof(float);   //  8 MB
  float*  vf   = (float*)ws;   ws += (size_t)S_LEN * KVDIM * sizeof(float);   //  8 MB
  bf16_t* qbf  = (bf16_t*)ws;  ws += (size_t)S_LEN * QDIM  * sizeof(bf16_t);  // 16 MB
  bf16_t* kbf  = (bf16_t*)ws;  ws += (size_t)S_LEN * KVDIM * sizeof(bf16_t);  //  4 MB
  bf16_t* vt   = (bf16_t*)ws;  ws += (size_t)S_LEN * KVDIM * sizeof(bf16_t);  //  4 MB
  bf16_t* atb  = (bf16_t*)ws;  ws += (size_t)S_LEN * QDIM  * sizeof(bf16_t);  // 16 MB

  float* out = (float*)d_out;  // [S, DIM] fp32

  // 0) one-time fp32 -> bf16 conversions
  {
    auto conv = [&](const float* src, bf16_t* dst, size_t n) {
      int n4 = (int)(n / 4);
      conv_f32_bf16<<<(n4 + 255) / 256, 256, 0, stream>>>(src, dst, n4);
    };
    conv(x,  xbf, (size_t)S_LEN * DIM);
    conv(wq, wqb, (size_t)QDIM * DIM);
    conv(wk, wkb, (size_t)KVDIM * DIM);
    conv(wv, wvb, (size_t)KVDIM * DIM);
    conv(wo, wob, (size_t)DIM * QDIM);
  }

  const dim3 blk_g(256);
  // 1) projections (bf16 x bf16 -> fp32)
  gemm_bf16_wmma<<<dim3(QDIM / 128,  S_LEN / 128), blk_g, 0, stream>>>(xbf, wqb, qf, S_LEN, QDIM,  DIM);
  gemm_bf16_wmma<<<dim3(KVDIM / 128, S_LEN / 128), blk_g, 0, stream>>>(xbf, wkb, kf, S_LEN, KVDIM, DIM);
  gemm_bf16_wmma<<<dim3(KVDIM / 128, S_LEN / 128), blk_g, 0, stream>>>(xbf, wvb, vf, S_LEN, KVDIM, DIM);

  // 2) RoPE + bf16 conversion (softmax scale folded into Q)
  const float scale = 0.08838834764831845f;  // 1/sqrt(128)
  {
    int nq = S_LEN * NH * (HD / 2);
    rope_conv_kernel<<<(nq + 255) / 256, 256, 0, stream>>>(qf, fcos, fsin, qbf, NH, scale);
    int nk = S_LEN * NKV * (HD / 2);
    rope_conv_kernel<<<(nk + 255) / 256, 256, 0, stream>>>(kf, fcos, fsin, kbf, NKV, 1.0f);
    int nv = S_LEN * NKV * HD;
    conv_v_kernel<<<(nv + 255) / 256, 256, 0, stream>>>(vf, vt);
  }

  // 3) flash attention (bf16 in, bf16 out)
  flash_kernel<<<dim3(S_LEN / 16, NH), dim3(32), 0, stream>>>(qbf, kbf, vt, atb);

  // 4) output projection
  gemm_bf16_wmma<<<dim3(DIM / 128, S_LEN / 128), blk_g, 0, stream>>>(atb, wob, out, S_LEN, DIM, QDIM);
}
